// CausalSelfAttention_79388175499619
// MI455X (gfx1250) — compile-verified, matrix operand order checked by reading, not run on hardware
//
#include <hip/hip_runtime.h>
#include <hip/hip_bf16.h>

#define B_  4
#define T_  2048
#define C_  1024
#define H_  16
#define D_  64
#define BT_ (B_ * T_)   // 8192

typedef __attribute__((ext_vector_type(16))) __bf16 v16bf;
typedef __attribute__((ext_vector_type(8)))  float  v8f;
typedef int v4i __attribute__((vector_size(16)));   // matches builtin's param type

union Frag16 { v16bf v; uint4 q[2]; };
union U4U16  { uint4 q; unsigned short s[8]; };

#if defined(__has_builtin)
#if __has_builtin(__builtin_amdgcn_global_load_async_to_lds_b128)
#define HAVE_ASYNC_LDS 1
#endif
#endif

#ifndef HAVE_ASYNC_LDS
#define HAVE_ASYNC_LDS 0
#endif

#if HAVE_ASYNC_LDS
#define AS1 __attribute__((address_space(1)))
#define AS3 __attribute__((address_space(3)))
__device__ __forceinline__ void async_copy_b128(const unsigned short* g, unsigned short* l) {
  __builtin_amdgcn_global_load_async_to_lds_b128((AS1 v4i*)g, (AS3 v4i*)l, 0, 0);
}
#define WAIT_ASYNC() asm volatile("s_wait_asynccnt 0" ::: "memory")
#else
#define WAIT_ASYNC()
#endif

__device__ __forceinline__ unsigned short f2bf(float x) {
  unsigned int u = __float_as_uint(x);
  u += 0x7fffu + ((u >> 16) & 1u);          // round-to-nearest-even
  return (unsigned short)(u >> 16);
}

// ---------------- f32 -> bf16 convert ----------------
__global__ __launch_bounds__(256) void cvt_bf16_kernel(const float* __restrict__ src,
                                                       unsigned short* __restrict__ dst,
                                                       int n) {
  int i = blockIdx.x * 256 + threadIdx.x;
  int stride = gridDim.x * 256;
  for (; i < n; i += stride) dst[i] = f2bf(src[i]);
}

// ---------------- bf16 GEMM:  out = A[8192xC] * W[CxC] + bias ----------------
// mode 0: out_bf head-shuffled [B,H,T,D] bf16      mode 1: out_f32 [BT,C] f32
__global__ __launch_bounds__(256) void gemm_bf16_kernel(
    const unsigned short* __restrict__ A,
    const unsigned short* __restrict__ W,
    const float* __restrict__ bias,
    unsigned short* __restrict__ out_bf,
    float* __restrict__ out_f32,
    int mode) {
  __shared__ __align__(16) unsigned short Abuf[2][128][40];  // ping-pong A tiles
  __shared__ __align__(16) unsigned short Btb[2][64][40];    // W tiles, transposed [n][k]

  const int m0   = blockIdx.x * 128;
  const int n0   = blockIdx.y * 64;
  const int tid  = threadIdx.x;
  const int lane = tid & 31;
  const int wave = tid >> 5;
  const int nloc = lane & 15;
  const int half = lane >> 4;
  const int koff = half * 8;                // A/B fragment K-run base per ISA layout

  // staging work split (fixed per thread)
  const int arow = tid >> 1;                // A: 128 rows x 32B
  const int ac0  = (tid & 1) * 16;
  const int bk   = tid >> 3;                // W: 32 k-rows x 64 n (8 bf16 per thread)
  const int bnn  = (tid & 7) * 8;

  v8f acc[4];
#pragma unroll
  for (int t = 0; t < 4; ++t)
#pragma unroll
    for (int r = 0; r < 8; ++r) acc[t][r] = 0.0f;

  auto stage = [&](int kc, int buf) {
    const unsigned short* ga = A + (size_t)(m0 + arow) * C_ + kc + ac0;
#if HAVE_ASYNC_LDS
    async_copy_b128(ga,     &Abuf[buf][arow][ac0]);        // ASYNCcnt path
    async_copy_b128(ga + 8, &Abuf[buf][arow][ac0 + 8]);
#else
    const uint4* g = (const uint4*)ga;
    *(uint4*)&Abuf[buf][arow][ac0]     = g[0];
    *(uint4*)&Abuf[buf][arow][ac0 + 8] = g[1];
#endif
    U4U16 wv;                                              // transpose W tile via DS
    wv.q = *(const uint4*)(W + (size_t)(kc + bk) * C_ + n0 + bnn);
#pragma unroll
    for (int i = 0; i < 8; ++i) Btb[buf][bnn + i][bk] = wv.s[i];
  };

  stage(0, 0);
  WAIT_ASYNC();
  __syncthreads();

  for (int kc = 0; kc < C_; kc += 32) {
    const int buf = (kc >> 5) & 1;
    if (kc + 32 < C_) stage(kc + 32, buf ^ 1);   // overlap next tile with WMMAs

    Frag16 af;   // A fragment: lane holds K runs [koff,koff+8) and [koff+16,koff+24)
    af.q[0] = *(const uint4*)&Abuf[buf][wave * 16 + nloc][koff];
    af.q[1] = *(const uint4*)&Abuf[buf][wave * 16 + nloc][koff + 16];
    Frag16 bfr[4];
#pragma unroll
    for (int nt = 0; nt < 4; ++nt) {
      bfr[nt].q[0] = *(const uint4*)&Btb[buf][nt * 16 + nloc][koff];
      bfr[nt].q[1] = *(const uint4*)&Btb[buf][nt * 16 + nloc][koff + 16];
    }
#pragma unroll
    for (int nt = 0; nt < 4; ++nt)
      acc[nt] = __builtin_amdgcn_wmma_f32_16x16x32_bf16(
          false, af.v, false, bfr[nt].v, (short)0, acc[nt], false, false);

    WAIT_ASYNC();
    __syncthreads();
  }

  // epilogue: C layout -> element (m = r + 8*half, n = nloc) per 16x16 tile
#pragma unroll
  for (int nt = 0; nt < 4; ++nt) {
#pragma unroll
    for (int r = 0; r < 8; ++r) {
      int mloc = r + half * 8;
      int m = m0 + wave * 16 + mloc;
      int n = n0 + nt * 16 + nloc;
      float val = acc[nt][r] + bias[n];
      if (mode == 0) {
        int b = m >> 11, t = m & (T_ - 1);
        int h = n >> 6,  d = n & 63;
        out_bf[((size_t)(b * H_ + h) * T_ + t) * D_ + d] = f2bf(val);
      } else {
        out_f32[(size_t)m * C_ + n] = val;
      }
    }
  }
}

// ---------------- flash attention: Q,K,V bf16 [B,H,T,D] -> O bf16 [B,T,C] ----------------
__global__ __launch_bounds__(256) void attn_kernel(
    const unsigned short* __restrict__ Q,
    const unsigned short* __restrict__ K,
    const unsigned short* __restrict__ V,
    unsigned short* __restrict__ O) {
  __shared__ __align__(16) unsigned short Kt[2][32][72];    // [key][d], ping-pong
  __shared__ __align__(16) unsigned short Vt[2][64][40];    // V transposed [d][key]
  __shared__ __align__(16) unsigned short Pscr[8][16][40];  // per-wave P relayout scratch

  const int bh   = blockIdx.x >> 4;          // b*H + h
  const int q0   = (blockIdx.x & 15) * 128;  // block query base
  const int tid  = threadIdx.x;
  const int lane = tid & 31;
  const int wave = tid >> 5;
  const int nloc = lane & 15;
  const int half = lane >> 4;
  const int koff = half * 8;
  const int qt   = q0 + wave * 16;           // this wave's query tile

  // staging work split
  const int skey = tid >> 3;
  const int sdd  = (tid & 7) * 8;

  auto stage = [&](int kb, int buf) {
    size_t base = ((size_t)bh * T_ + kb + skey) * D_ + sdd;
#if HAVE_ASYNC_LDS
    async_copy_b128(K + base, &Kt[buf][skey][sdd]);
#else
    *(uint4*)&Kt[buf][skey][sdd] = *(const uint4*)(K + base);
#endif
    U4U16 vv;
    vv.q = *(const uint4*)(V + base);
#pragma unroll
    for (int i = 0; i < 8; ++i) Vt[buf][sdd + i][skey] = vv.s[i];
  };

  // Q fragments (A layout), two K=32 chunks over D=64 — loaded once, reused all key blocks
  Frag16 qf[2];
  {
    const unsigned short* qrow = Q + ((size_t)bh * T_ + (qt + nloc)) * D_;
#pragma unroll
    for (int c = 0; c < 2; ++c) {
      qf[c].q[0] = *(const uint4*)(qrow + c * 32 + koff);
      qf[c].q[1] = *(const uint4*)(qrow + c * 32 + koff + 16);
    }
  }

  v8f o_[4];
#pragma unroll
  for (int t = 0; t < 4; ++t)
#pragma unroll
    for (int r = 0; r < 8; ++r) o_[t][r] = 0.0f;
  float mrun[8], lrun[8];
#pragma unroll
  for (int r = 0; r < 8; ++r) { mrun[r] = -__builtin_inff(); lrun[r] = 0.0f; }

  const int kmax = q0 + 128;                 // causal: keys needed by this block

  stage(0, 0);
  WAIT_ASYNC();
  __syncthreads();

  for (int kb = 0; kb < kmax; kb += 32) {
    const int buf = (kb >> 5) & 1;
    if (kb + 32 < kmax) stage(kb + 32, buf ^ 1);  // prefetch next K/V tile

    if (kb <= qt + 15) {                     // causal range for this wave
      // ---- S = Q * K^T  (two 16x16 score tiles, K-dim = D in 2 chunks) ----
      Frag16 kf[2][2];
#pragma unroll
      for (int c = 0; c < 2; ++c) {
        kf[c][0].q[0] = *(const uint4*)&Kt[buf][nloc][c * 32 + koff];
        kf[c][0].q[1] = *(const uint4*)&Kt[buf][nloc][c * 32 + koff + 16];
        kf[c][1].q[0] = *(const uint4*)&Kt[buf][16 + nloc][c * 32 + koff];
        kf[c][1].q[1] = *(const uint4*)&Kt[buf][16 + nloc][c * 32 + koff + 16];
      }
      v8f s0, s1;
#pragma unroll
      for (int r = 0; r < 8; ++r) { s0[r] = 0.0f; s1[r] = 0.0f; }
#pragma unroll
      for (int c = 0; c < 2; ++c) {
        s0 = __builtin_amdgcn_wmma_f32_16x16x32_bf16(false, qf[c].v, false, kf[c][0].v,
                                                     (short)0, s0, false, false);
        s1 = __builtin_amdgcn_wmma_f32_16x16x32_bf16(false, qf[c].v, false, kf[c][1].v,
                                                     (short)0, s1, false, false);
      }

      // ---- causal mask + online softmax (rows live in 16-lane halves) ----
#pragma unroll
      for (int r = 0; r < 8; ++r) {
        int mloc = r + half * 8;
        int qg   = qt + mloc;
        float v0 = s0[r] * 0.125f;             // 1/sqrt(64)
        float v1 = s1[r] * 0.125f;
        if (kb + nloc > qg)      v0 = -__builtin_inff();
        if (kb + 16 + nloc > qg) v1 = -__builtin_inff();
        float mx = fmaxf(v0, v1);
#pragma unroll
        for (int off = 1; off < 16; off <<= 1)
          mx = fmaxf(mx, __shfl_xor(mx, off, 32));
        float mnew  = fmaxf(mrun[r], mx);
        float alpha = __expf(mrun[r] - mnew);
        mrun[r] = mnew;
        float p0 = __expf(v0 - mnew);
        float p1 = __expf(v1 - mnew);
        float rs = p0 + p1;
#pragma unroll
        for (int off = 1; off < 16; off <<= 1)
          rs += __shfl_xor(rs, off, 32);
        lrun[r] = lrun[r] * alpha + rs;
#pragma unroll
        for (int nt = 0; nt < 4; ++nt) o_[nt][r] *= alpha;
        Pscr[wave][mloc][nloc]      = f2bf(p0);   // relayout C -> A via LDS
        Pscr[wave][mloc][16 + nloc] = f2bf(p1);
      }

      asm volatile("s_wait_dscnt 0" ::: "memory");  // wave-local scratch RAW

      // ---- O += P * V ----
      Frag16 pf;
      pf.q[0] = *(const uint4*)&Pscr[wave][nloc][koff];
      pf.q[1] = *(const uint4*)&Pscr[wave][nloc][koff + 16];
      Frag16 vf[4];
#pragma unroll
      for (int nt = 0; nt < 4; ++nt) {
        vf[nt].q[0] = *(const uint4*)&Vt[buf][nt * 16 + nloc][koff];
        vf[nt].q[1] = *(const uint4*)&Vt[buf][nt * 16 + nloc][koff + 16];
      }
#pragma unroll
      for (int nt = 0; nt < 4; ++nt)
        o_[nt] = __builtin_amdgcn_wmma_f32_16x16x32_bf16(
            false, pf.v, false, vf[nt].v, (short)0, o_[nt], false, false);
    }

    WAIT_ASYNC();
    __syncthreads();                          // uniform across all waves
  }

  // ---- normalize + store bf16 [B,T,C] ----
  const int b = bh >> 4;
  const int h = bh & 15;
#pragma unroll
  for (int r = 0; r < 8; ++r) {
    int mloc = r + half * 8;
    float inv = 1.0f / lrun[r];
    size_t row = (size_t)(b * T_ + qt + mloc) * C_ + h * D_;
#pragma unroll
    for (int nt = 0; nt < 4; ++nt)
      O[row + nt * 16 + nloc] = f2bf(o_[nt][r] * inv);
  }
}

extern "C" void kernel_launch(void* const* d_in, const int* in_sizes, int n_in,
                              void* d_out, int out_size, void* d_ws, size_t ws_size,
                              hipStream_t stream) {
  const float* x  = (const float*)d_in[0];
  const float* Wq = (const float*)d_in[1];
  const float* bq = (const float*)d_in[2];
  const float* Wk = (const float*)d_in[3];
  const float* bk = (const float*)d_in[4];
  const float* Wv = (const float*)d_in[5];
  const float* bv = (const float*)d_in[6];
  const float* Wo = (const float*)d_in[7];
  const float* bo = (const float*)d_in[8];
  float* out = (float*)d_out;

  unsigned char* ws = (unsigned char*)d_ws;
  size_t off = 0;
  auto alloc = [&](size_t bytes) -> void* {
    void* p = ws + off;
    off += (bytes + 255) & ~(size_t)255;
    return p;
  };
  const size_t actBytes = (size_t)BT_ * C_ * 2;  // 16 MB bf16
  const size_t wBytes   = (size_t)C_ * C_ * 2;   // 2 MB bf16
  unsigned short* xbf  = (unsigned short*)alloc(actBytes);
  unsigned short* wqbf = (unsigned short*)alloc(wBytes);
  unsigned short* wkbf = (unsigned short*)alloc(wBytes);
  unsigned short* wvbf = (unsigned short*)alloc(wBytes);
  unsigned short* wobf = (unsigned short*)alloc(wBytes);
  unsigned short* qbf  = (unsigned short*)alloc(actBytes);  // [B,H,T,D]
  unsigned short* kbf  = (unsigned short*)alloc(actBytes);
  unsigned short* vbf  = (unsigned short*)alloc(actBytes);
  unsigned short* ybf  = (unsigned short*)alloc(actBytes);  // [B,T,C]

  const int nX = BT_ * C_;
  const int nW = C_ * C_;
  cvt_bf16_kernel<<<4096, 256, 0, stream>>>(x,  xbf,  nX);
  cvt_bf16_kernel<<<1024, 256, 0, stream>>>(Wq, wqbf, nW);
  cvt_bf16_kernel<<<1024, 256, 0, stream>>>(Wk, wkbf, nW);
  cvt_bf16_kernel<<<1024, 256, 0, stream>>>(Wv, wvbf, nW);
  cvt_bf16_kernel<<<1024, 256, 0, stream>>>(Wo, wobf, nW);

  dim3 ggrid(BT_ / 128, C_ / 64);
  gemm_bf16_kernel<<<ggrid, 256, 0, stream>>>(xbf, wqbf, bq, qbf, nullptr, 0);
  gemm_bf16_kernel<<<ggrid, 256, 0, stream>>>(xbf, wkbf, bk, kbf, nullptr, 0);
  gemm_bf16_kernel<<<ggrid, 256, 0, stream>>>(xbf, wvbf, bv, vbf, nullptr, 0);

  attn_kernel<<<B_ * H_ * (T_ / 128), 256, 0, stream>>>(qbf, kbf, vbf, ybf);

  gemm_bf16_kernel<<<ggrid, 256, 0, stream>>>(ybf, wobf, bo, nullptr, out, 1);
}
